// GIN_54228257079641
// MI455X (gfx1250) — matrix operand order repaired and checked by reading, not compiled, not run on hardware
//
#include <hip/hip_runtime.h>

typedef unsigned short u16;
typedef __attribute__((ext_vector_type(16))) __bf16    v16bf;
typedef __attribute__((ext_vector_type(8)))  float     v8f;
typedef __attribute__((ext_vector_type(4)))  unsigned  v4u;

union Frag16 { v16bf bf; v4u q[2]; };

#define Nn   100000
#define Ee   640000
#define Ll   3
#define Hh   128
#define Gg   512
#define OUTc 16
#define ROWS 64
#define LSTR 136   // 128 + 8 u16 pad -> 272B row stride, staggers LDS banks

__device__ __forceinline__ u16 f2bf(float f) {
  unsigned u = __float_as_uint(f);
  u += 0x7FFFu + ((u >> 16) & 1u);   // round-to-nearest-even
  return (u16)(u >> 16);
}

// ---- pack fp32 weight matrix (K x Ncols, row-major) into WMMA B-fragment order:
// out[((colTile*kSteps + kStep)*32 + lane)*16 + j] = bf16(W[k][n])
//   n = colTile*16 + (lane&15), k = kStep*32 + (j>>3)*16 + (lane>>4)*8 + (j&7)
__global__ void pack_w(const float* __restrict__ W, u16* __restrict__ out, int K, int Ncols) {
  int id = blockIdx.x * 256 + threadIdx.x;
  int total = K * Ncols;
  if (id >= total) return;
  int j      = id & 15;
  int lane   = (id >> 4) & 31;
  int rest   = id >> 9;
  int kSteps = K >> 5;
  int kStep  = rest % kSteps;
  int colTile = rest / kSteps;
  int n = colTile * 16 + (lane & 15);
  int k = kStep * 32 + (j >> 3) * 16 + (lane >> 4) * 8 + (j & 7);
  out[id] = f2bf(W[(size_t)k * Ncols + n]);
}

__global__ void zero_f32(float* __restrict__ p, int n) {
  int t = blockIdx.x * 256 + threadIdx.x;
  if (t < n) p[t] = 0.f;
}

// z = (1 + eps[l]) * hin   (vectorized float4)
__global__ __launch_bounds__(256) void scale_nodes(const float* __restrict__ hin,
                                                   float* __restrict__ z,
                                                   const float* __restrict__ eps,
                                                   int l, int n4) {
  int t = blockIdx.x * 256 + threadIdx.x;
  if (t >= n4) return;
  float s = 1.0f + eps[l];
  float4 v = reinterpret_cast<const float4*>(hin)[t];
  v.x *= s; v.y *= s; v.z *= s; v.w *= s;
  reinterpret_cast<float4*>(z)[t] = v;
}

// one wave per edge: lane handles 4 contiguous floats; scatter-add into z[dst]
__global__ __launch_bounds__(256) void edge_agg(const float* __restrict__ h,
                                                const int* __restrict__ ei,
                                                float* __restrict__ z, int ne) {
  int t = blockIdx.x * 256 + threadIdx.x;
  int e = t >> 5;
  int lane = t & 31;
  if (e >= ne) return;
  int src = ei[e];
  int dst = ei[ne + e];
  float4 v = *reinterpret_cast<const float4*>(h + (size_t)src * Hh + lane * 4);
  float* zp = z + (size_t)dst * Hh + lane * 4;
  atomicAdd(zp + 0, v.x);
  atomicAdd(zp + 1, v.y);
  atomicAdd(zp + 2, v.z);
  atomicAdd(zp + 3, v.w);
}

// Fused per-layer MLP: h = relu( relu(bn(relu(z@W1+b1))) @ W2 + b2 );
// also pooled[batch[i]] += h[i].  64 rows per block, 8 waves = 8 column tiles.
__global__ __launch_bounds__(256) void gin_mlp_kernel(
    const float* __restrict__ zin,
    const u16* __restrict__ W1p, const float* __restrict__ b1,
    const float* __restrict__ gm, const float* __restrict__ bt,
    const float* __restrict__ mu, const float* __restrict__ vr,
    const u16* __restrict__ W2p, const float* __restrict__ b2,
    float* __restrict__ hout, float* __restrict__ pooled,
    const int* __restrict__ batch, int nrows) {
  __shared__ __align__(16) u16 As[ROWS][LSTR];
  __shared__ __align__(16) u16 Z1[ROWS][LSTR];
  const int tid  = threadIdx.x;
  const int lane = tid & 31;
  const int wave = tid >> 5;
  const int half = lane >> 4;
  const int m    = lane & 15;
  const int row0 = blockIdx.x * ROWS;

  // stage A block (f32 -> bf16) into LDS
  for (int it = tid; it < ROWS * (Hh / 4); it += 256) {
    int r = it >> 5, c4 = it & 31, g = row0 + r;
    float4 v = make_float4(0.f, 0.f, 0.f, 0.f);
    if (g < nrows) v = *reinterpret_cast<const float4*>(zin + (size_t)g * Hh + c4 * 4);
    As[r][c4 * 4 + 0] = f2bf(v.x);
    As[r][c4 * 4 + 1] = f2bf(v.y);
    As[r][c4 * 4 + 2] = f2bf(v.z);
    As[r][c4 * 4 + 3] = f2bf(v.w);
  }
  __syncthreads();

  const int colTile = wave;
  const int col = colTile * 16 + m;

  Frag16 B[4];
#pragma unroll
  for (int ks = 0; ks < 4; ++ks) {
    const v4u* p = reinterpret_cast<const v4u*>(W1p + ((size_t)(colTile * 4 + ks) * 32 + lane) * 16);
    B[ks].q[0] = p[0]; B[ks].q[1] = p[1];
  }
  float bias1 = b1[col];
  float sc = gm[col] * rsqrtf(vr[col] + 1e-5f);
  float sh = bt[col] - mu[col] * sc;

#pragma unroll
  for (int rt = 0; rt < 4; ++rt) {
    v8f acc = {};
#pragma unroll
    for (int ks = 0; ks < 4; ++ks) {
      Frag16 A;
      A.q[0] = *reinterpret_cast<const v4u*>(&As[rt * 16 + m][ks * 32 + half * 8]);
      A.q[1] = *reinterpret_cast<const v4u*>(&As[rt * 16 + m][ks * 32 + 16 + half * 8]);
      acc = __builtin_amdgcn_wmma_f32_16x16x32_bf16(false, A.bf, false, B[ks].bf,
                                                    (short)0, acc, false, false);
    }
#pragma unroll
    for (int r = 0; r < 8; ++r) {
      float v = acc[r] + bias1;
      v = v > 0.f ? v : 0.f;          // ReLU
      v = v * sc + sh;                // BatchNorm (eval)
      v = v > 0.f ? v : 0.f;          // ReLU
      Z1[rt * 16 + half * 8 + r][col] = f2bf(v);
    }
  }
  __syncthreads();

#pragma unroll
  for (int ks = 0; ks < 4; ++ks) {
    const v4u* p = reinterpret_cast<const v4u*>(W2p + ((size_t)(colTile * 4 + ks) * 32 + lane) * 16);
    B[ks].q[0] = p[0]; B[ks].q[1] = p[1];
  }
  float bias2 = b2[col];

#pragma unroll
  for (int rt = 0; rt < 4; ++rt) {
    v8f acc = {};
#pragma unroll
    for (int ks = 0; ks < 4; ++ks) {
      Frag16 A;
      A.q[0] = *reinterpret_cast<const v4u*>(&Z1[rt * 16 + m][ks * 32 + half * 8]);
      A.q[1] = *reinterpret_cast<const v4u*>(&Z1[rt * 16 + m][ks * 32 + 16 + half * 8]);
      acc = __builtin_amdgcn_wmma_f32_16x16x32_bf16(false, A.bf, false, B[ks].bf,
                                                    (short)0, acc, false, false);
    }
#pragma unroll
    for (int r = 0; r < 8; ++r) {
      int g = row0 + rt * 16 + half * 8 + r;
      if (g < nrows) {
        float v = acc[r] + bias2;
        v = v > 0.f ? v : 0.f;        // outer GIN ReLU
        hout[(size_t)g * Hh + col] = v;
        atomicAdd(&pooled[(size_t)batch[g] * Hh + col], v);
      }
    }
  }
}

// Head: out = relu(P @ lin1 + b1) @ lin2 + b2   (P = pooled[2], 512x128; OUT=16)
__global__ __launch_bounds__(256) void gin_head_kernel(
    const float* __restrict__ P, const u16* __restrict__ L1p,
    const float* __restrict__ l1b, const u16* __restrict__ L2p,
    const float* __restrict__ l2b, float* __restrict__ out) {
  __shared__ __align__(16) u16 As[ROWS][LSTR];
  __shared__ __align__(16) u16 Z1[ROWS][LSTR];
  const int tid = threadIdx.x;
  const int lane = tid & 31, wave = tid >> 5;
  const int half = lane >> 4, m = lane & 15;
  const int row0 = blockIdx.x * ROWS;

  for (int it = tid; it < ROWS * (Hh / 4); it += 256) {
    int r = it >> 5, c4 = it & 31;
    float4 v = *reinterpret_cast<const float4*>(P + (size_t)(row0 + r) * Hh + c4 * 4);
    As[r][c4 * 4 + 0] = f2bf(v.x);
    As[r][c4 * 4 + 1] = f2bf(v.y);
    As[r][c4 * 4 + 2] = f2bf(v.z);
    As[r][c4 * 4 + 3] = f2bf(v.w);
  }
  __syncthreads();

  const int colTile = wave;
  const int col = colTile * 16 + m;
  Frag16 B[4];
#pragma unroll
  for (int ks = 0; ks < 4; ++ks) {
    const v4u* p = reinterpret_cast<const v4u*>(L1p + ((size_t)(colTile * 4 + ks) * 32 + lane) * 16);
    B[ks].q[0] = p[0]; B[ks].q[1] = p[1];
  }
  float bias1 = l1b[col];
#pragma unroll
  for (int rt = 0; rt < 4; ++rt) {
    v8f acc = {};
#pragma unroll
    for (int ks = 0; ks < 4; ++ks) {
      Frag16 A;
      A.q[0] = *reinterpret_cast<const v4u*>(&As[rt * 16 + m][ks * 32 + half * 8]);
      A.q[1] = *reinterpret_cast<const v4u*>(&As[rt * 16 + m][ks * 32 + 16 + half * 8]);
      acc = __builtin_amdgcn_wmma_f32_16x16x32_bf16(false, A.bf, false, B[ks].bf,
                                                    (short)0, acc, false, false);
    }
#pragma unroll
    for (int r = 0; r < 8; ++r) {
      float v = acc[r] + bias1;
      v = v > 0.f ? v : 0.f;
      Z1[rt * 16 + half * 8 + r][col] = f2bf(v);
    }
  }
  __syncthreads();

  if (wave < 4) {    // OUT=16 -> single column tile; one wave per row tile
    Frag16 B2[4];
#pragma unroll
    for (int ks = 0; ks < 4; ++ks) {
      const v4u* p = reinterpret_cast<const v4u*>(L2p + ((size_t)ks * 32 + lane) * 16);
      B2[ks].q[0] = p[0]; B2[ks].q[1] = p[1];
    }
    const int rt = wave;
    v8f acc = {};
#pragma unroll
    for (int ks = 0; ks < 4; ++ks) {
      Frag16 A;
      A.q[0] = *reinterpret_cast<const v4u*>(&Z1[rt * 16 + m][ks * 32 + half * 8]);
      A.q[1] = *reinterpret_cast<const v4u*>(&Z1[rt * 16 + m][ks * 32 + 16 + half * 8]);
      acc = __builtin_amdgcn_wmma_f32_16x16x32_bf16(false, A.bf, false, B2[ks].bf,
                                                    (short)0, acc, false, false);
    }
    float bias2 = l2b[m];
#pragma unroll
    for (int r = 0; r < 8; ++r) {
      out[(size_t)(row0 + rt * 16 + half * 8 + r) * OUTc + m] = acc[r] + bias2;
    }
  }
}

extern "C" void kernel_launch(void* const* d_in, const int* in_sizes, int n_in,
                              void* d_out, int out_size, void* d_ws, size_t ws_size,
                              hipStream_t stream) {
  const float* x     = (const float*)d_in[0];
  const int*   ei    = (const int*)d_in[1];
  const int*   batch = (const int*)d_in[2];
  const float* eps   = (const float*)d_in[3];
  const float* W1    = (const float*)d_in[4];
  const float* b1    = (const float*)d_in[5];
  const float* gm    = (const float*)d_in[6];
  const float* bt    = (const float*)d_in[7];
  const float* mu    = (const float*)d_in[8];
  const float* vr    = (const float*)d_in[9];
  const float* W2    = (const float*)d_in[10];
  const float* b2    = (const float*)d_in[11];
  const float* l1w   = (const float*)d_in[12];
  const float* l1b   = (const float*)d_in[13];
  const float* l2w   = (const float*)d_in[14];
  const float* l2b   = (const float*)d_in[15];

  float* out = (float*)d_out;
  float* pooledBase = out + Gg * OUTc;   // pooled[0..2] live in d_out

  // workspace: h (N*H f32) | z (N*H f32) | packed bf16 weights
  float* hbuf = (float*)d_ws;
  float* zbuf = hbuf + (size_t)Nn * Hh;
  u16* W1p = (u16*)(zbuf + (size_t)Nn * Hh);
  u16* W2p = W1p + (size_t)Ll * Hh * Hh;
  u16* L1p = W2p + (size_t)Ll * Hh * Hh;
  u16* L2p = L1p + (size_t)Hh * Hh;

  const int totW = Hh * Hh, gbW = (totW + 255) / 256;
  for (int l = 0; l < Ll; ++l) {
    pack_w<<<gbW, 256, 0, stream>>>(W1 + (size_t)l * Hh * Hh, W1p + (size_t)l * Hh * Hh, Hh, Hh);
    pack_w<<<gbW, 256, 0, stream>>>(W2 + (size_t)l * Hh * Hh, W2p + (size_t)l * Hh * Hh, Hh, Hh);
  }
  pack_w<<<gbW, 256, 0, stream>>>(l1w, L1p, Hh, Hh);
  pack_w<<<(Hh * OUTc + 255) / 256, 256, 0, stream>>>(l2w, L2p, Hh, OUTc);

  zero_f32<<<(Ll * Gg * Hh + 255) / 256, 256, 0, stream>>>(pooledBase, Ll * Gg * Hh);

  const int n4 = Nn * Hh / 4;
  const int mlpBlocks = (Nn + ROWS - 1) / ROWS;
  for (int l = 0; l < Ll; ++l) {
    const float* hin = (l == 0) ? x : hbuf;
    scale_nodes<<<(n4 + 255) / 256, 256, 0, stream>>>(hin, zbuf, eps, l, n4);
    edge_agg<<<(Ee * 32) / 256, 256, 0, stream>>>(hin, ei, zbuf, Ee);
    gin_mlp_kernel<<<mlpBlocks, 256, 0, stream>>>(
        zbuf,
        W1p + (size_t)l * Hh * Hh, b1 + l * Hh, gm + l * Hh, bt + l * Hh,
        mu + l * Hh, vr + l * Hh,
        W2p + (size_t)l * Hh * Hh, b2 + l * Hh,
        hbuf, pooledBase + (size_t)l * Gg * Hh, batch, Nn);
  }
  gin_head_kernel<<<Gg / ROWS, 256, 0, stream>>>(
      pooledBase + (size_t)2 * Gg * Hh, L1p, l1b, L2p, l2b, out);
}
